// ClrSelfAttention_76355928588566
// MI455X (gfx1250) — matrix operand order repaired and checked by reading, not compiled, not run on hardware
//
#include <hip/hip_runtime.h>

// ---------------------------------------------------------------------------
// Types for CDNA5 WMMA (gfx1250, wave32)
// ---------------------------------------------------------------------------
typedef __attribute__((ext_vector_type(16))) __bf16 v16bf;
typedef __attribute__((ext_vector_type(8)))  float  v8f;
typedef __attribute__((ext_vector_type(8)))  unsigned int v8u;

typedef unsigned short bf16_t; // storage type for bf16 (avoid __bf16 scalar ops)

__device__ __forceinline__ unsigned short f2bf(float x) {
    unsigned int u = __float_as_uint(x);
    unsigned int r = u + 0x7fffu + ((u >> 16) & 1u); // round-to-nearest-even
    return (unsigned short)(r >> 16);
}

// CDNA5 async global->LDS copy (GLOBAL_LOAD_ASYNC_TO_LDS_B128, ASYNCcnt).
// ldsoff = per-lane LDS byte address, gaddr = per-lane 64-bit global address.
__device__ __forceinline__ void async_copy_b128(unsigned int ldsoff, const void* gaddr) {
    asm volatile("global_load_async_to_lds_b128 %0, %1, off"
                 :: "v"(ldsoff), "v"((unsigned long long)gaddr) : "memory");
}
__device__ __forceinline__ void wait_asynccnt0() {
    asm volatile("s_wait_asynccnt 0x0" ::: "memory");
}
__device__ __forceinline__ unsigned int lds_offset(const void* p) {
    return (unsigned int)(unsigned long long)p; // LDS aperture: addr[31:0] = LDS offset
}

// constants
#define BATCH 2
#define SEQT  2048
#define DMODEL 1024
#define NHEAD 16
#define DHEAD 64
#define NROWS (BATCH * SEQT)   // 4096

// ---------------------------------------------------------------------------
// Block reductions (256 threads)
// ---------------------------------------------------------------------------
__device__ __forceinline__ float block_sum(float v, float* red) {
    int tid = threadIdx.x;
    red[tid] = v; __syncthreads();
    for (int off = 128; off > 0; off >>= 1) {
        if (tid < off) red[tid] += red[tid + off];
        __syncthreads();
    }
    float r = red[0]; __syncthreads();
    return r;
}
__device__ __forceinline__ float block_max(float v, float* red) {
    int tid = threadIdx.x;
    red[tid] = v; __syncthreads();
    for (int off = 128; off > 0; off >>= 1) {
        if (tid < off) red[tid] = fmaxf(red[tid], red[tid + off]);
        __syncthreads();
    }
    float r = red[0]; __syncthreads();
    return r;
}

// ---------------------------------------------------------------------------
// Kernel 1: z = log(p+eps) - rowmean  -> bf16 [4096][1024]
// ---------------------------------------------------------------------------
__global__ __launch_bounds__(256) void clr_kernel(const float* __restrict__ p,
                                                  bf16_t* __restrict__ z) {
    __shared__ float red[256];
    int row = blockIdx.x;
    const float* pr = p + (size_t)row * DMODEL;
    bf16_t* zr = z + (size_t)row * DMODEL;
    float v[4]; float s = 0.f;
    for (int i = 0; i < 4; ++i) {
        float x = __logf(pr[threadIdx.x + 256 * i] + 1e-6f);
        v[i] = x; s += x;
    }
    float mean = block_sum(s, red) * (1.0f / DMODEL);
    for (int i = 0; i < 4; ++i)
        zr[threadIdx.x + 256 * i] = f2bf(v[i] - mean);
}

// ---------------------------------------------------------------------------
// Kernel 2: weight transpose f32 [K=1024][N=1024] -> bf16 Wt[N][K]
// ---------------------------------------------------------------------------
__global__ __launch_bounds__(256) void transpose_w_kernel(const float* __restrict__ W,
                                                          bf16_t* __restrict__ Wt) {
    __shared__ float tile[32][33];
    int bx = blockIdx.x * 32; // n
    int by = blockIdx.y * 32; // k
    int tx = threadIdx.x, ty = threadIdx.y; // (32,8)
    for (int i = 0; i < 32; i += 8)
        tile[ty + i][tx] = W[(size_t)(by + ty + i) * DMODEL + bx + tx];
    __syncthreads();
    for (int i = 0; i < 32; i += 8)
        Wt[(size_t)(bx + ty + i) * DMODEL + by + tx] = f2bf(tile[tx][ty + i]);
}

// ---------------------------------------------------------------------------
// Kernel 3: GEMM  C[M][1024] (f32) = A_bf16[M][1024] @ B (+bias),
//           B supplied transposed: Wt[n][k].
// 256 threads = 8 waves; block tile 64x128; each wave 32x32 (2x2 WMMA tiles).
// Double-buffered LDS filled with GLOBAL_LOAD_ASYNC_TO_LDS_B128: the async
// copy engine fills the next K-tile while the wave issues WMMAs; drained with
// a single s_wait_asynccnt 0 before the per-step barrier.
// ---------------------------------------------------------------------------
__global__ __launch_bounds__(256) void gemm_bias_kernel(const bf16_t* __restrict__ A,
                                                        const bf16_t* __restrict__ Wt,
                                                        const float* __restrict__ bias,
                                                        float* __restrict__ C) {
    const int K = DMODEL, N = DMODEL;
    const int LDA = 40; // halfword LDS stride (pad; 80B rows keep 16B alignment)
    __shared__ __align__(16) bf16_t As[2][64 * 40];
    __shared__ __align__(16) bf16_t Bs[2][128 * 40];

    int tid = threadIdx.x;
    int lane = tid & 31, wave = tid >> 5;
    int half = lane >> 4, l15 = lane & 15;
    int wm = wave & 1, wn = wave >> 1;
    int m0 = blockIdx.y * 64, n0 = blockIdx.x * 128;

    unsigned int ldsA = lds_offset(&As[0][0]);
    unsigned int ldsB = lds_offset(&Bs[0][0]);

    // async tile fill: A = 1 b128 chunk/thread, B = 2 chunks/thread
    auto load_tile = [&](int bsel, int k0) {
        {
            int r = tid >> 2, cw = tid & 3;            // 256 chunks = 64 rows x 4
            async_copy_b128(ldsA + (unsigned int)(bsel * 64 * 40 + r * LDA + cw * 8) * 2u,
                            A + (size_t)(m0 + r) * K + k0 + cw * 8);
        }
        {
            int r = tid >> 2, cw = tid & 3;            // chunks 0..255
            async_copy_b128(ldsB + (unsigned int)(bsel * 128 * 40 + r * LDA + cw * 8) * 2u,
                            Wt + (size_t)(n0 + r) * K + k0 + cw * 8);
        }
        {
            int c = tid + 256;                         // chunks 256..511
            int r = c >> 2, cw = c & 3;
            async_copy_b128(ldsB + (unsigned int)(bsel * 128 * 40 + r * LDA + cw * 8) * 2u,
                            Wt + (size_t)(n0 + r) * K + k0 + cw * 8);
        }
    };

    // A-fragment K offsets per ISA 16-bit A 16x32 layout
    int akoff[8];
    for (int j = 0; j < 8; ++j)
        akoff[j] = (j < 4) ? (half * 8 + 2 * j) : (16 + half * 8 + 2 * (j - 4));

    v8f acc[2][2] = {};

    load_tile(0, 0);
    wait_asynccnt0();
    __syncthreads();

    int buf = 0;
    for (int k0 = 0; k0 < K; k0 += 32) {
        int nbuf = buf ^ 1;
        if (k0 + 32 < K) load_tile(nbuf, k0 + 32); // async, overlaps WMMAs below

        v16bf af[2], bfv[2];
        for (int mi = 0; mi < 2; ++mi) {
            v8u t;
            const bf16_t* base = &As[buf][0] + (wm * 32 + mi * 16 + l15) * LDA;
            for (int j = 0; j < 8; ++j)
                t[j] = *(const unsigned int*)(base + akoff[j]);
            af[mi] = __builtin_bit_cast(v16bf, t);
        }
        for (int ni = 0; ni < 2; ++ni) {
            v8u t;
            const bf16_t* base = &Bs[buf][0] + (wn * 32 + ni * 16 + l15) * LDA;
            for (int j = 0; j < 8; ++j)
                t[j] = *(const unsigned int*)(base + akoff[j]);
            bfv[ni] = __builtin_bit_cast(v16bf, t);
        }
        for (int mi = 0; mi < 2; ++mi)
            for (int ni = 0; ni < 2; ++ni)
                acc[mi][ni] = __builtin_amdgcn_wmma_f32_16x16x32_bf16(
                    false, af[mi], false, bfv[ni], (short)0, acc[mi][ni], false, false);

        wait_asynccnt0();  // this wave's async copies into nbuf have landed
        __syncthreads();   // all waves' copies landed; buf reads done
        buf = nbuf;
    }

    // epilogue: + bias, f32 out (C/D layout: lanes<16 rows r, lanes>=16 rows r+8)
    for (int mi = 0; mi < 2; ++mi)
        for (int ni = 0; ni < 2; ++ni) {
            int col = n0 + wn * 32 + ni * 16 + l15;
            float b = bias[col];
            for (int r = 0; r < 8; ++r) {
                int row = m0 + wm * 32 + mi * 16 + r + half * 8;
                C[(size_t)row * N + col] = acc[mi][ni][r] + b;
            }
        }
}

// ---------------------------------------------------------------------------
// Kernel 4: QKV postprocess: proj_H over row, convert bf16, repack.
//  transposeV==0: dst[b][h][t][d]      (for Q, K)
//  transposeV==1: dst[b][h][d][t]      (for V, feeds P@V B-fragments)
// ---------------------------------------------------------------------------
__global__ __launch_bounds__(256) void qkv_post_kernel(const float* __restrict__ raw,
                                                       bf16_t* __restrict__ dst,
                                                       int transposeV) {
    __shared__ float red[256];
    int row = blockIdx.x;              // b*T + t
    int b = row >> 11, t = row & (SEQT - 1);
    const float* rr = raw + (size_t)row * DMODEL;
    float v[4]; float s = 0.f;
    for (int i = 0; i < 4; ++i) { v[i] = rr[threadIdx.x + 256 * i]; s += v[i]; }
    float mean = block_sum(s, red) * (1.0f / DMODEL);
    for (int i = 0; i < 4; ++i) {
        int col = threadIdx.x + 256 * i;
        int h = col >> 6, d = col & 63;
        size_t bh = (size_t)(b * NHEAD + h);
        size_t idx = transposeV ? ((bh * DHEAD + d) * SEQT + t)
                                : ((bh * SEQT + t) * DHEAD + d);
        dst[idx] = f2bf(v[i] - mean);
    }
}

// ---------------------------------------------------------------------------
// Kernel 5: flash attention. 8 waves/block, one wave = 16 query rows of one
// (b,h). Streams keys in steps of 32: S = Q K^T (4 WMMA), online softmax,
// O += P V (4 WMMA). P bounced via per-wave LDS slab (C-layout -> A-fragment
// layout); same-wave LDS ops are in-order on CDNA5, so a wave-local
// s_wait_dscnt (not a block barrier) is all the ordering we need.
// ---------------------------------------------------------------------------
__global__ __launch_bounds__(256) void flash_attn_kernel(const bf16_t* __restrict__ Q,
                                                         const bf16_t* __restrict__ Kh,
                                                         const bf16_t* __restrict__ Vt,
                                                         bf16_t* __restrict__ O) {
    __shared__ __align__(16) bf16_t Pld[8][16 * 32];
    int tid = threadIdx.x, lane = tid & 31, wave = tid >> 5;
    int half = lane >> 4, l15 = lane & 15;
    int bh = blockIdx.x >> 4;                        // 0..31
    int i0 = (blockIdx.x & 15) * 128 + wave * 16;    // query row base
    int b = bh >> 4, h = bh & 15;

    const bf16_t* Qb = Q  + (size_t)bh * SEQT * DHEAD;
    const bf16_t* Kb = Kh + (size_t)bh * SEQT * DHEAD;
    const bf16_t* Vb = Vt + (size_t)bh * DHEAD * SEQT;

    int akoff[8];
    for (int j = 0; j < 8; ++j)
        akoff[j] = (j < 4) ? (half * 8 + 2 * j) : (16 + half * 8 + 2 * (j - 4));

    // Q fragments (K = 0..31 and 32..63)
    v16bf qf[2];
    for (int kf = 0; kf < 2; ++kf) {
        v8u t;
        const bf16_t* base = Qb + (size_t)(i0 + l15) * DHEAD + kf * 32;
        for (int j = 0; j < 8; ++j) t[j] = *(const unsigned int*)(base + akoff[j]);
        qf[kf] = __builtin_bit_cast(v16bf, t);
    }

    float mrun[8], rsum[8];
    for (int r = 0; r < 8; ++r) { mrun[r] = -1e30f; rsum[r] = 0.f; }
    v8f oacc[4] = {};
    bf16_t* myP = &Pld[wave][0];

    for (int j0 = 0; j0 < SEQT; j0 += 32) {
        // prefetch next key/value tiles into cache (global_prefetch_b8)
        if (j0 + 32 < SEQT) {
            __builtin_prefetch(Kb + (size_t)(j0 + 32 + l15) * DHEAD, 0, 1);
            __builtin_prefetch(Kb + (size_t)(j0 + 48 + l15) * DHEAD, 0, 1);
            for (int nt = 0; nt < 4; ++nt)
                __builtin_prefetch(Vb + (size_t)(nt * 16 + l15) * SEQT + j0 + 32, 0, 1);
        }
        // ---- S = Q K^T for two 16-wide key tiles
        v8f s[2] = {};
        for (int jt = 0; jt < 2; ++jt) {
            int jrow = j0 + jt * 16 + l15;
            const bf16_t* kbase = Kb + (size_t)jrow * DHEAD;
            for (int kf = 0; kf < 2; ++kf) {
                v8u t;
                for (int j = 0; j < 8; ++j)
                    t[j] = *(const unsigned int*)(kbase + kf * 32 + akoff[j]);
                v16bf kfrag = __builtin_bit_cast(v16bf, t);
                s[jt] = __builtin_amdgcn_wmma_f32_16x16x32_bf16(
                    false, qf[kf], false, kfrag, (short)0, s[jt], false, false);
            }
        }
        // ---- online softmax (rows split across lane halves)
        for (int r = 0; r < 8; ++r) {
            float s0 = s[0][r] * 0.125f; // 1/sqrt(64)
            float s1 = s[1][r] * 0.125f;
            float mx = fmaxf(s0, s1);
            for (int msk = 1; msk < 16; msk <<= 1)
                mx = fmaxf(mx, __shfl_xor(mx, msk, 32));
            float mnew = fmaxf(mrun[r], mx);
            float alpha = __expf(mrun[r] - mnew);
            mrun[r] = mnew;
            float p0 = __expf(s0 - mnew);
            float p1 = __expf(s1 - mnew);
            float rs = p0 + p1;
            for (int msk = 1; msk < 16; msk <<= 1)
                rs += __shfl_xor(rs, msk, 32);
            rsum[r] = rsum[r] * alpha + rs;
            for (int nt = 0; nt < 4; ++nt) oacc[nt][r] *= alpha;
            int grow = r + half * 8;
            myP[grow * 32 + l15]      = f2bf(p0);
            myP[grow * 32 + 16 + l15] = f2bf(p1);
        }
        // wave-local LDS fence: drain our u16 stores, block TBAA reordering
        asm volatile("s_wait_dscnt 0x0" ::: "memory");
        // ---- P as A-fragment, O += P @ V
        v8u t;
        for (int j = 0; j < 8; ++j)
            t[j] = *(const unsigned int*)(myP + l15 * 32 + akoff[j]);
        v16bf pf = __builtin_bit_cast(v16bf, t);
        for (int nt = 0; nt < 4; ++nt) {
            v8u tv;
            int d = nt * 16 + l15;
            const bf16_t* vbase = Vb + (size_t)d * SEQT + j0;
            for (int j = 0; j < 8; ++j)
                tv[j] = *(const unsigned int*)(vbase + akoff[j]);
            v16bf vfrag = __builtin_bit_cast(v16bf, tv);
            oacc[nt] = __builtin_amdgcn_wmma_f32_16x16x32_bf16(
                false, pf, false, vfrag, (short)0, oacc[nt], false, false);
        }
        // compiler-only fence: keep next iter's P stores after this iter's loads
        asm volatile("" ::: "memory");
    }

    // normalize and write token-major bf16 [b][t][h*64+d]
    for (int nt = 0; nt < 4; ++nt) {
        int d = nt * 16 + l15;
        for (int r = 0; r < 8; ++r) {
            int trow = i0 + r + half * 8;
            float val = oacc[nt][r] / rsum[r];
            O[((size_t)(b * SEQT + trow)) * DMODEL + h * DHEAD + d] = f2bf(val);
        }
    }
}

// ---------------------------------------------------------------------------
// Kernel 6: proj_H + softmax over D -> f32 out
// ---------------------------------------------------------------------------
__global__ __launch_bounds__(256) void proj_softmax_kernel(const float* __restrict__ raw,
                                                           float* __restrict__ out) {
    __shared__ float red[256];
    int row = blockIdx.x;
    const float* rr = raw + (size_t)row * DMODEL;
    float* orow = out + (size_t)row * DMODEL;
    float v[4]; float s = 0.f;
    for (int i = 0; i < 4; ++i) { v[i] = rr[threadIdx.x + 256 * i]; s += v[i]; }
    float mean = block_sum(s, red) * (1.0f / DMODEL);
    float mx = -1e30f;
    for (int i = 0; i < 4; ++i) { v[i] -= mean; mx = fmaxf(mx, v[i]); }
    mx = block_max(mx, red);
    float es = 0.f;
    for (int i = 0; i < 4; ++i) { v[i] = __expf(v[i] - mx); es += v[i]; }
    float denom = block_sum(es, red);
    float inv = 1.0f / denom;
    for (int i = 0; i < 4; ++i) orow[threadIdx.x + 256 * i] = v[i] * inv;
}

// ---------------------------------------------------------------------------
// Host launcher
// ---------------------------------------------------------------------------
extern "C" void kernel_launch(void* const* d_in, const int* in_sizes, int n_in,
                              void* d_out, int out_size, void* d_ws, size_t ws_size,
                              hipStream_t stream) {
    (void)in_sizes; (void)n_in; (void)out_size; (void)ws_size;
    const float* p  = (const float*)d_in[0];
    const float* Wq = (const float*)d_in[1];
    const float* bq = (const float*)d_in[2];
    const float* Wk = (const float*)d_in[3];
    const float* bk = (const float*)d_in[4];
    const float* Wv = (const float*)d_in[5];
    const float* bv = (const float*)d_in[6];
    const float* Wo = (const float*)d_in[7];
    const float* bo = (const float*)d_in[8];
    float* out = (float*)d_out;

    char* ws = (char*)d_ws;
    const size_t MB = 1ull << 20;
    bf16_t* zbf  = (bf16_t*)(ws + 0);        // 8 MB  [4096][1024] bf16
    bf16_t* Wq_t = (bf16_t*)(ws + 8 * MB);   // 2 MB each
    bf16_t* Wk_t = (bf16_t*)(ws + 10 * MB);
    bf16_t* Wv_t = (bf16_t*)(ws + 12 * MB);
    bf16_t* Wo_t = (bf16_t*)(ws + 14 * MB);
    float*  raw  = (float*)(ws + 16 * MB);   // 16 MB [4096][1024] f32 (reused 4x)
    bf16_t* qh   = (bf16_t*)(ws + 32 * MB);  // 8 MB  [B][NH][T][DK]
    bf16_t* kh   = (bf16_t*)(ws + 40 * MB);  // 8 MB
    bf16_t* vT   = (bf16_t*)(ws + 48 * MB);  // 8 MB  [B][NH][DK][T]
    bf16_t* ao   = zbf;                      // reuse (z dead after QKV GEMMs)

    clr_kernel<<<NROWS, 256, 0, stream>>>(p, zbf);

    dim3 tb(32, 8), tg(32, 32);
    transpose_w_kernel<<<tg, tb, 0, stream>>>(Wq, Wq_t);
    transpose_w_kernel<<<tg, tb, 0, stream>>>(Wk, Wk_t);
    transpose_w_kernel<<<tg, tb, 0, stream>>>(Wv, Wv_t);
    transpose_w_kernel<<<tg, tb, 0, stream>>>(Wo, Wo_t);

    dim3 gg(DMODEL / 128, NROWS / 64); // (8, 64)
    gemm_bias_kernel<<<gg, 256, 0, stream>>>(zbf, Wq_t, bq, raw);
    qkv_post_kernel<<<NROWS, 256, 0, stream>>>(raw, qh, 0);
    gemm_bias_kernel<<<gg, 256, 0, stream>>>(zbf, Wk_t, bk, raw);
    qkv_post_kernel<<<NROWS, 256, 0, stream>>>(raw, kh, 0);
    gemm_bias_kernel<<<gg, 256, 0, stream>>>(zbf, Wv_t, bv, raw);
    qkv_post_kernel<<<NROWS, 256, 0, stream>>>(raw, vT, 1);

    flash_attn_kernel<<<BATCH * NHEAD * (SEQT / 128), 256, 0, stream>>>(qh, kh, vT, ao);

    gemm_bias_kernel<<<gg, 256, 0, stream>>>(ao, Wo_t, bo, raw);
    proj_softmax_kernel<<<NROWS, 256, 0, stream>>>(raw, out);
}
